// FDAttention_21131239096376
// MI455X (gfx1250) — compile-verified, matrix operand order checked by reading, not run on hardware
//
#include <hip/hip_runtime.h>

typedef __attribute__((ext_vector_type(16))) _Float16 v16h;
typedef __attribute__((ext_vector_type(8)))  float    v8f;

#define BATCH  8
#define NTOK   4096
#define DIM    512
#define NQKV   1536
#define HEADS  8
#define DHEAD  64
#define H2     4
#define NT     16
#define SPOS   256
#define QKV_ELEMS ((size_t)BATCH * HEADS * NTOK * DHEAD)   // 16777216 per tensor
#define TOFF      ((size_t)BATCH * H2 * NTOK * DHEAD)      // temporal out offset (floats)

union Frag {
    v16h h;
    unsigned u[8];
    _Float16 e[16];
};

// K-pair base for 16-bit A/B fragment VGPR vg, lane-half g (ISA 7.12.2):
// VGPR0..3: K = {0,1},{2,3},{4,5},{6,7} (+8 for lanes 16-31)
// VGPR4..7: K = {16,17}..{22,23}        (+8 for lanes 16-31)
__device__ __forceinline__ int kpair(int vg, int g) {
    return (vg < 4) ? (2 * vg + 8 * g) : (8 + 2 * vg + 8 * g);
}

// Async global->LDS copy, 16 bytes per lane (ASYNCcnt-tracked; ISA 10.x / 08_async_tensor).
// lds_off: per-lane LDS byte address (low 32 bits of a generic shared pointer).
__device__ __forceinline__ void async_copy_b128(unsigned lds_off, const void* gaddr) {
    asm volatile("global_load_async_to_lds_b128 %0, %1, off"
                 :: "v"(lds_off), "v"(gaddr)
                 : "memory");
}
__device__ __forceinline__ void wait_async0() {
    asm volatile("s_wait_asynccnt 0x0" ::: "memory");
}

// ---------------------------------------------------------------------------
// Kernel 1: qkv = x @ w_qkv  (M=32768, K=512, N=1536), f16 WMMA, f32 accum.
// Block: 256 thr = 8 waves; tile 64(M) x 64(N); waves = 4(M) x 2(N), each wave
// computes 16x32 via two 16x16x32 WMMAs per k-step.
// Epilogue scatters into head-major q/k/v f16 buffers; q scaled by 1/8.
// ---------------------------------------------------------------------------
__global__ __launch_bounds__(256) void qkv_gemm(const float* __restrict__ x,
                                                const float* __restrict__ w,
                                                _Float16* __restrict__ qb,
                                                _Float16* __restrict__ kb,
                                                _Float16* __restrict__ vb) {
    __shared__ __align__(16) _Float16 As[64][34];   // [m][k]
    __shared__ __align__(16) _Float16 Bt[64][34];   // [n][k] (transposed)

    const int tid  = threadIdx.x;
    const int lane = tid & 31, wid = tid >> 5;
    const int g = lane >> 4, r = lane & 15;
    const int wm = wid & 3, wn = wid >> 2;
    const int m0 = blockIdx.y * 64;
    const int n0 = blockIdx.x * 64;

    v8f acc0 = {}, acc1 = {};

    for (int kk = 0; kk < DIM; kk += 32) {
        if (kk + 32 < DIM) {   // prefetch next tiles -> global_prefetch_b8
            __builtin_prefetch(&x[(size_t)(m0 + (tid >> 2)) * DIM + kk + 32], 0, 1);
            __builtin_prefetch(&w[(size_t)(kk + 32 + (tid >> 3)) * NQKV + n0], 0, 1);
        }
#pragma unroll
        for (int i = 0; i < 8; ++i) {               // A: 64x32 f32 -> f16
            int e = i * 256 + tid;
            int row = e >> 5, col = e & 31;
            As[row][col] = (_Float16)x[(size_t)(m0 + row) * DIM + kk + col];
        }
#pragma unroll
        for (int i = 0; i < 8; ++i) {               // B: 32x64, store transposed
            int e = i * 256 + tid;
            int kr = e >> 6, nc = e & 63;
            Bt[nc][kr] = (_Float16)w[(size_t)(kk + kr) * NQKV + n0 + nc];
        }
        __syncthreads();

        Frag a, b0, b1;
#pragma unroll
        for (int vg = 0; vg < 8; ++vg) {
            int kb_ = kpair(vg, g);
            a.u[vg]  = *(const unsigned*)&As[wm * 16 + r][kb_];
            b0.u[vg] = *(const unsigned*)&Bt[wn * 32 + r][kb_];
            b1.u[vg] = *(const unsigned*)&Bt[wn * 32 + 16 + r][kb_];
        }
        acc0 = __builtin_amdgcn_wmma_f32_16x16x32_f16(false, a.h, false, b0.h,
                                                      (short)0, acc0, false, false);
        acc1 = __builtin_amdgcn_wmma_f32_16x16x32_f16(false, a.h, false, b1.h,
                                                      (short)0, acc1, false, false);
        __syncthreads();
    }

    const int which = n0 >> 9;            // 0=q 1=k 2=v (64-col tile never crosses)
    const int head  = (n0 & 511) >> 6;
    _Float16* dst = (which == 0) ? qb : (which == 1 ? kb : vb);
    const float scl = (which == 0) ? 0.125f : 1.0f;   // fold d^-0.5 into Q
#pragma unroll
    for (int nt = 0; nt < 2; ++nt) {
        v8f acc = nt ? acc1 : acc0;
        int dcol = wn * 32 + nt * 16 + r;
#pragma unroll
        for (int vg = 0; vg < 8; ++vg) {
            int m = m0 + wm * 16 + vg + 8 * g;
            int bidx = m >> 12, token = m & 4095;
            size_t idx = ((size_t)((bidx * HEADS + head) * NTOK + token)) * DHEAD + dcol;
            dst[idx] = (_Float16)(acc[vg] * scl);
        }
    }
}

// ---------------------------------------------------------------------------
// Kernel 2: spatial attention. 1024 blocks; block = half of one
// (b, head<4, frame) problem; 8 waves x 16 query rows. Flash-style online
// softmax over 8 key chunks of 32. K/V tiles are staged with double-buffered
// GLOBAL_LOAD_ASYNC_TO_LDS_B128 (ASYNCcnt), overlapping next-chunk loads with
// the current chunk's WMMAs.
// ---------------------------------------------------------------------------
__global__ __launch_bounds__(256) void spatial_attn(const _Float16* __restrict__ qb,
                                                    const _Float16* __restrict__ kb,
                                                    const _Float16* __restrict__ vb,
                                                    float* __restrict__ out) {
    // 72-half pitch (144 B) keeps every per-lane B128 LDS destination 16B-aligned.
    __shared__ __align__(16) _Float16 Ks[2][32][72];     // [buf][j][d] row-major
    __shared__ __align__(16) _Float16 Vr[2][32][72];     // [buf][j][d] row-major
    __shared__ __align__(16) _Float16 Pb[8][16][34];     // per-wave P tile

    const int tid  = threadIdx.x;
    const int lane = tid & 31, wid = tid >> 5;
    const int g = lane >> 4, r = lane & 15;

    const int blk  = blockIdx.x;
    const int half = blk & 1;
    const int prob = blk >> 1;               // 512 = b*4*16
    const int b = prob >> 6;
    const int h = (prob >> 4) & 3;
    const int t = prob & 15;

    const size_t headbase = (size_t)(b * HEADS + h) * NTOK;   // heads 0..3
    const int tok0 = t * SPOS + half * 128 + wid * 16;

    // per-thread async-copy coordinates: 256 thr x 16B = one 32x64 f16 tile
    const int cj = tid >> 3;          // row 0..31
    const int cd = (tid & 7) * 8;     // 8-half (16B) chunk

    Frag aq[2];
    {
        const _Float16* qrow = qb + (headbase + tok0 + r) * DHEAD;
#pragma unroll
        for (int c = 0; c < 2; ++c)
#pragma unroll
            for (int vg = 0; vg < 8; ++vg)
                aq[c].u[vg] = *(const unsigned*)&qrow[c * 32 + kpair(vg, g)];
    }

    float mi[8], li[8];
    v8f o0 = {}, o1 = {}, o2 = {}, o3 = {};
#pragma unroll
    for (int i = 0; i < 8; ++i) { mi[i] = -1e30f; li[i] = 0.f; }

    // kick off chunk 0 into buffer 0
    {
        const size_t grow = (headbase + t * SPOS + cj) * DHEAD + cd;
        async_copy_b128((unsigned)(size_t)&Ks[0][cj][cd], kb + grow);
        async_copy_b128((unsigned)(size_t)&Vr[0][cj][cd], vb + grow);
    }

    for (int jc = 0; jc < 8; ++jc) {
        const int cur = jc & 1;
        wait_async0();          // this wave's pending tile copies done
        __syncthreads();        // all waves' copies visible; prev buffer free

        if (jc + 1 < 8) {       // prefetch next chunk into the other buffer
            const size_t grow = (headbase + t * SPOS + (jc + 1) * 32 + cj) * DHEAD + cd;
            async_copy_b128((unsigned)(size_t)&Ks[1 - cur][cj][cd], kb + grow);
            async_copy_b128((unsigned)(size_t)&Vr[1 - cur][cj][cd], vb + grow);
        }

        v8f s0 = {}, s1 = {};
#pragma unroll
        for (int c = 0; c < 2; ++c) {        // S = Q K^T over d
            Frag bk0, bk1;
#pragma unroll
            for (int vg = 0; vg < 8; ++vg) {
                int kb_ = c * 32 + kpair(vg, g);
                bk0.u[vg] = *(const unsigned*)&Ks[cur][r][kb_];
                bk1.u[vg] = *(const unsigned*)&Ks[cur][16 + r][kb_];
            }
            s0 = __builtin_amdgcn_wmma_f32_16x16x32_f16(false, aq[c].h, false, bk0.h,
                                                        (short)0, s0, false, false);
            s1 = __builtin_amdgcn_wmma_f32_16x16x32_f16(false, aq[c].h, false, bk1.h,
                                                        (short)0, s1, false, false);
        }

#pragma unroll
        for (int vg = 0; vg < 8; ++vg) {     // online softmax, row m = vg+8g
            float a = s0[vg], bv2 = s1[vg];
            float mx = fmaxf(a, bv2);
#pragma unroll
            for (int off = 8; off > 0; off >>= 1)
                mx = fmaxf(mx, __shfl_xor(mx, off, 32));
            float mnew  = fmaxf(mi[vg], mx);
            float alpha = __expf(mi[vg] - mnew);
            float p0 = __expf(a - mnew);
            float p1 = __expf(bv2 - mnew);
            float rs = p0 + p1;
#pragma unroll
            for (int off = 8; off > 0; off >>= 1)
                rs += __shfl_xor(rs, off, 32);
            li[vg] = li[vg] * alpha + rs;
            mi[vg] = mnew;
            o0[vg] *= alpha; o1[vg] *= alpha; o2[vg] *= alpha; o3[vg] *= alpha;
            int row = vg + 8 * g;
            Pb[wid][row][r]      = (_Float16)p0;
            Pb[wid][row][16 + r] = (_Float16)p1;
        }

        Frag ap;                              // P (C-layout) -> A fragment via LDS
#pragma unroll
        for (int vg = 0; vg < 8; ++vg)
            ap.u[vg] = *(const unsigned*)&Pb[wid][r][kpair(vg, g)];
#pragma unroll
        for (int c4 = 0; c4 < 4; ++c4) {      // O += P V  (V row-major: paired u16 loads)
            Frag bv;
#pragma unroll
            for (int vg = 0; vg < 8; ++vg) {
                int kb_ = kpair(vg, g);
                bv.e[2 * vg]     = Vr[cur][kb_][c4 * 16 + r];
                bv.e[2 * vg + 1] = Vr[cur][kb_ + 1][c4 * 16 + r];
            }
            v8f& oo = (c4 == 0) ? o0 : (c4 == 1) ? o1 : (c4 == 2) ? o2 : o3;
            oo = __builtin_amdgcn_wmma_f32_16x16x32_f16(false, ap.h, false, bv.h,
                                                        (short)0, oo, false, false);
        }
    }

#pragma unroll
    for (int vg = 0; vg < 8; ++vg) {
        float inv = 1.0f / li[vg];
        int token = tok0 + vg + 8 * g;
        size_t base = ((size_t)((b * H2 + h) * NTOK + token)) * DHEAD;
        out[base +  0 + r] = o0[vg] * inv;
        out[base + 16 + r] = o1[vg] * inv;
        out[base + 32 + r] = o2[vg] * inv;
        out[base + 48 + r] = o3[vg] * inv;
    }
}

// ---------------------------------------------------------------------------
// Kernel 3: temporal attention. Reference's raw reshape makes temporal tokens
// contiguous: n = sp*16 + t. One wave per (b, head 4..7, sp) 16x16x64 problem;
// P.V WMMA zero-padded to K=32 in the upper fragment halves.
// ---------------------------------------------------------------------------
__global__ __launch_bounds__(256) void temporal_attn(const _Float16* __restrict__ qb,
                                                     const _Float16* __restrict__ kb,
                                                     const _Float16* __restrict__ vb,
                                                     float* __restrict__ out) {
    __shared__ __align__(16) _Float16 Pb[8][16][18];
    __shared__ __align__(16) _Float16 Vt[8][64][18];

    const int tid  = threadIdx.x;
    const int lane = tid & 31, wid = tid >> 5;
    const int g = lane >> 4, r = lane & 15;

    const int prob = blockIdx.x * 8 + wid;          // 8192 = b*4*256
    const int b  = prob >> 10;
    const int h  = (prob >> 8) & 3;
    const int sp = prob & 255;
    const size_t rowbase = ((size_t)((b * HEADS + 4 + h) * NTOK + sp * 16)) * DHEAD;

    Frag aq[2];
#pragma unroll
    for (int c = 0; c < 2; ++c)
#pragma unroll
        for (int vg = 0; vg < 8; ++vg)
            aq[c].u[vg] = *(const unsigned*)&qb[rowbase + (size_t)r * DHEAD + c * 32 + kpair(vg, g)];

#pragma unroll
    for (int i = 0; i < 32; ++i) {                  // stage V transposed (per wave)
        int e = i * 32 + lane;
        int j = e >> 6, d = e & 63;
        Vt[wid][d][j] = vb[rowbase + j * DHEAD + d];
    }

    v8f s = {};
#pragma unroll
    for (int c = 0; c < 2; ++c) {                   // S = Q K^T (K frag direct from global)
        Frag bk;
#pragma unroll
        for (int vg = 0; vg < 8; ++vg)
            bk.u[vg] = *(const unsigned*)&kb[rowbase + (size_t)r * DHEAD + c * 32 + kpair(vg, g)];
        s = __builtin_amdgcn_wmma_f32_16x16x32_f16(false, aq[c].h, false, bk.h,
                                                   (short)0, s, false, false);
    }

    float li[8];
#pragma unroll
    for (int vg = 0; vg < 8; ++vg) {                // full softmax over 16 cols
        float mx = s[vg];
#pragma unroll
        for (int off = 8; off > 0; off >>= 1)
            mx = fmaxf(mx, __shfl_xor(mx, off, 32));
        float p  = __expf(s[vg] - mx);
        float rs = p;
#pragma unroll
        for (int off = 8; off > 0; off >>= 1)
            rs += __shfl_xor(rs, off, 32);
        li[vg] = rs;
        Pb[wid][vg + 8 * g][r] = (_Float16)p;
    }

    Frag ap;                                        // A = [P | 0]  (K padded 16->32)
#pragma unroll
    for (int vg = 0; vg < 8; ++vg)
        ap.u[vg] = (vg < 4) ? *(const unsigned*)&Pb[wid][r][kpair(vg, g)] : 0u;

    v8f o[4] = {v8f{}, v8f{}, v8f{}, v8f{}};
#pragma unroll
    for (int c4 = 0; c4 < 4; ++c4) {
        Frag bv;
#pragma unroll
        for (int vg = 0; vg < 8; ++vg)
            bv.u[vg] = (vg < 4) ? *(const unsigned*)&Vt[wid][c4 * 16 + r][kpair(vg, g)] : 0u;
        o[c4] = __builtin_amdgcn_wmma_f32_16x16x32_f16(false, ap.h, false, bv.h,
                                                       (short)0, o[c4], false, false);
    }

#pragma unroll
    for (int vg = 0; vg < 8; ++vg) {
        float inv = 1.0f / li[vg];
        int m = vg + 8 * g;
        size_t base = TOFF + ((size_t)((b * H2 + h) * NTOK + sp * 16 + m)) * DHEAD;
        out[base +  0 + r] = o[0][vg] * inv;
        out[base + 16 + r] = o[1][vg] * inv;
        out[base + 32 + r] = o[2][vg] * inv;
        out[base + 48 + r] = o[3][vg] * inv;
    }
}

extern "C" void kernel_launch(void* const* d_in, const int* in_sizes, int n_in,
                              void* d_out, int out_size, void* d_ws, size_t ws_size,
                              hipStream_t stream) {
    const float* x = (const float*)d_in[0];
    const float* w = (const float*)d_in[1];

    _Float16* qb = (_Float16*)d_ws;
    _Float16* kb = (_Float16*)((char*)d_ws + QKV_ELEMS * sizeof(_Float16));
    _Float16* vb = (_Float16*)((char*)d_ws + 2 * QKV_ELEMS * sizeof(_Float16));
    float* out = (float*)d_out;

    // 1) qkv projection: M=32768 x N=1536 x K=512, 64x64 tiles
    qkv_gemm<<<dim3(NQKV / 64, (BATCH * NTOK) / 64), 256, 0, stream>>>(x, w, qb, kb, vb);
    // 2) spatial attention: 512 problems x 2 half-blocks
    spatial_attn<<<dim3(1024), 256, 0, stream>>>(qb, kb, vb, out);
    // 3) temporal attention: 8192 problems / 8 waves per block
    temporal_attn<<<dim3(1024), 256, 0, stream>>>(qb, kb, vb, out);
}
